// AttentionDecoderRNN_10952166605179
// MI455X (gfx1250) — compile-verified, matrix operand order checked by reading, not run on hardware
//
#include <hip/hip_runtime.h>
#include <hip/hip_bf16.h>

// ---------------------------------------------------------------------------
// Problem constants (B=64, S=64, T=32, H=512, V=32000)
// ---------------------------------------------------------------------------
#define B_  64
#define S_  64
#define T_  32
#define H_  512
#define V_  32000

typedef __attribute__((ext_vector_type(16))) __bf16 v16bf;
typedef __attribute__((ext_vector_type(8)))  float  v8f;

// ---------------------------------------------------------------------------
// WMMA fragment loaders (CDNA5 16x16x32 bf16 layouts, ISA 7.12.2):
//  A (16x32, MxK): lane L holds row M=L&15; K = kb+e (e<8), kb+16+(e-8) (e>=8),
//  with kb = (L>=16 ? 8 : 0).  B (32x16, KxN) mirrors A with N per lane.
// ---------------------------------------------------------------------------
__device__ __forceinline__ v16bf load_tile_f32(const float* __restrict__ base,
                                               int ld, int row0, int k0, int lane) {
  const float* p = base + (size_t)(row0 + (lane & 15)) * ld + k0 + ((lane >> 4) << 3);
  v16bf r;
#pragma unroll
  for (int e = 0; e < 8; ++e) { r[e] = (__bf16)p[e]; r[8 + e] = (__bf16)p[16 + e]; }
  return r;
}

__device__ __forceinline__ v16bf load_tile_bf16(const __bf16* __restrict__ base,
                                                int ld, int row0, int k0, int lane) {
  const __bf16* p = base + (size_t)(row0 + (lane & 15)) * ld + k0 + ((lane >> 4) << 3);
  v16bf r;
#pragma unroll
  for (int e = 0; e < 8; ++e) { r[e] = p[e]; r[8 + e] = p[16 + e]; }
  return r;
}

// ---------------------------------------------------------------------------
// Generic C = A(f32, MxK) @ W(bf16, NxK)^T + bias ; one 16x16 tile per wave.
// Used for the small per-step GEMMs (q, gi, gh) and Ua_keys.
// ---------------------------------------------------------------------------
__global__ void gemm_bf16w(const float* __restrict__ A, const __bf16* __restrict__ W,
                           const float* __restrict__ bias, float* __restrict__ C,
                           int M, int N, int K) {
  int wave = (int)((blockIdx.x * blockDim.x + threadIdx.x) >> 5);
  int lane = threadIdx.x & 31;
  int ntn = N >> 4;
  int ntiles = (M >> 4) * ntn;
  if (wave >= ntiles) return;
  int tm = wave / ntn, tn = wave % ntn;

  v8f acc = {};
  for (int k0 = 0; k0 < K; k0 += 32) {
    v16bf a = load_tile_f32(A, K, tm * 16, k0, lane);
    v16bf b = load_tile_bf16(W, K, tn * 16, k0, lane);
    acc = __builtin_amdgcn_wmma_f32_16x16x32_bf16(false, a, false, b,
                                                  (short)0, acc, false, false);
  }
  int n = tn * 16 + (lane & 15);
  float bv = bias ? bias[n] : 0.0f;
  int mbase = tm * 16 + ((lane >> 4) << 3);   // D: VGPR i -> M = i or 8+i
#pragma unroll
  for (int i = 0; i < 8; ++i) C[(size_t)(mbase + i) * N + n] = acc[i] + bv;
}

// ---------------------------------------------------------------------------
// Hoisted logits GEMM: out[(b*T+t), v] = Hall_bf[(t*B+b), :] @ Wo_bf^T + bo.
//  - A row-panel (16 x 512 bf16 = 16 KB) staged into LDS once per block via
//    CDNA5 async-to-LDS DMA (global_load_async_to_lds_b128 + s_wait_asynccnt),
//    then shared by all 8 waves through ds_load.
//  - 5 N-tiles register-blocked per wave: 50 blocks x 8 waves x 5 tiles = 2000
//    column tiles exactly (no partial waves -> EXEC all-ones for WMMA).
// ---------------------------------------------------------------------------
__global__ void __launch_bounds__(256) logits_kernel(
    const __bf16* __restrict__ hallbf, const __bf16* __restrict__ Wo,
    const float* __restrict__ bo, float* __restrict__ out) {
  constexpr int NTW = 5;                       // N-tiles per wave
  __shared__ __align__(16) __bf16 aSh[16 * H_];

  int tid  = threadIdx.x;
  int wave = tid >> 5, lane = tid & 31;
  int tm   = blockIdx.x / 50;                  // 128 row tiles
  int nblk = blockIdx.x % 50;                  // 50 column blocks (40 tiles each)

  // ---- async DMA stage of the A row-panel into LDS (ASYNCcnt path) ----
  {
    unsigned long long g =
        (unsigned long long)((const char*)(hallbf + (size_t)tm * 16 * H_) + tid * 64);
    unsigned lds = (unsigned)(uintptr_t)(&aSh[0]) + (unsigned)tid * 64;
#pragma unroll
    for (int i = 0; i < 4; ++i) {
      unsigned la = lds + i * 16;
      unsigned long long ga = g + i * 16;
      asm volatile("global_load_async_to_lds_b128 %0, %1, off"
                   :: "v"(la), "v"(ga) : "memory");
    }
    asm volatile("s_wait_asynccnt 0x0" ::: "memory");
  }
  __syncthreads();

  int tn0 = (nblk * 8 + wave) * NTW;           // first of this wave's 5 n-tiles
  __builtin_prefetch(Wo + (size_t)(tn0 * 16 + (lane & 15)) * H_, 0, 1);

  int arow = lane & 15;
  int kb   = (lane >> 4) << 3;
  v8f acc[NTW] = {};
  for (int k0 = 0; k0 < H_; k0 += 32) {
    const __bf16* ap = &aSh[arow * H_ + k0 + kb];
    v16bf a;
#pragma unroll
    for (int e = 0; e < 8; ++e) { a[e] = ap[e]; a[8 + e] = ap[16 + e]; }
#pragma unroll
    for (int j = 0; j < NTW; ++j) {
      v16bf b = load_tile_bf16(Wo, H_, (tn0 + j) * 16, k0, lane);
      acc[j] = __builtin_amdgcn_wmma_f32_16x16x32_bf16(false, a, false, b,
                                                       (short)0, acc[j], false, false);
    }
  }

  int mbase = tm * 16 + kb;
#pragma unroll
  for (int j = 0; j < NTW; ++j) {
    int n = (tn0 + j) * 16 + (lane & 15);
    float bias = bo[n];
#pragma unroll
    for (int i = 0; i < 8; ++i) {
      int r = mbase + i;                       // r = t*B + b
      int t = r >> 6, b = r & (B_ - 1);
      out[(size_t)(b * T_ + t) * V_ + n] = acc[j][i] + bias;
    }
  }
}

// ---------------------------------------------------------------------------
// Small helper kernels
// ---------------------------------------------------------------------------
__global__ void f32_to_bf16(const float* __restrict__ s, __bf16* __restrict__ d, int n) {
  int i = blockIdx.x * blockDim.x + threadIdx.x;
  if (i < n) d[i] = (__bf16)s[i];
}

__global__ void copy_f32(const float* __restrict__ s, float* __restrict__ d, int n) {
  int i = blockIdx.x * blockDim.x + threadIdx.x;
  if (i < n) d[i] = s[i];
}

// embedded[b,t,:] = emb[input_ids[b,t],:], input_ids = [SOS, target[:, :-1]]
__global__ void gather_emb(const int* __restrict__ tgt, const float* __restrict__ emb,
                           float* __restrict__ out) {
  int bt = blockIdx.x;                 // b*T + t
  int b = bt / T_, t = bt % T_;
  int id = (t == 0) ? 0 : tgt[b * T_ + (t - 1)];
  const float* s = emb + (size_t)id * H_;
  float* d = out + (size_t)bt * H_;
  for (int i = threadIdx.x; i < H_; i += blockDim.x) d[i] = s[i];
}

// Additive attention for one step: scores -> softmax -> context; also assembles
// x = [emb_t ; context] for the GRU input GEMM.  One block per batch row.
__global__ void attn_kernel(const float* __restrict__ q, const float* __restrict__ keys,
                            const float* __restrict__ va, const float* __restrict__ bvp,
                            const float* __restrict__ enc, const float* __restrict__ embedded,
                            float* __restrict__ xbuf, float* __restrict__ attn_out, int t) {
  int b = blockIdx.x;
  int tid = threadIdx.x;               // 256 threads = 8 waves
  int wave = tid >> 5, lane = tid & 31;
  __shared__ float sc[S_];
  __shared__ float red[2];

  const float* qq = q + (size_t)b * H_;
#pragma unroll
  for (int si = 0; si < S_ / 8; ++si) {
    int s = wave * (S_ / 8) + si;
    const float* kk = keys + (size_t)(b * S_ + s) * H_;
    float p = 0.0f;
    for (int h = lane; h < H_; h += 32) p += va[h] * tanhf(qq[h] + kk[h]);
    for (int off = 16; off; off >>= 1) p += __shfl_xor(p, off, 32);
    if (lane == 0) sc[s] = p + bvp[0];
  }
  __syncthreads();
  if (tid == 0) {
    float mx = sc[0];
    for (int s = 1; s < S_; ++s) mx = fmaxf(mx, sc[s]);
    red[0] = mx;
  }
  __syncthreads();
  float mx = red[0];
  if (tid < S_) sc[tid] = __expf(sc[tid] - mx);
  __syncthreads();
  if (tid == 0) {
    float sm = 0.0f;
    for (int s = 0; s < S_; ++s) sm += sc[s];
    red[1] = 1.0f / sm;
  }
  __syncthreads();
  float inv = red[1];
  if (tid < S_) attn_out[(size_t)(b * T_ + t) * S_ + tid] = sc[tid] * inv;
  __syncthreads();

  for (int h = tid; h < H_; h += blockDim.x) {
    float acc = 0.0f;
    for (int s = 0; s < S_; ++s) acc += sc[s] * enc[(size_t)(b * S_ + s) * H_ + h];
    xbuf[(size_t)b * (2 * H_) + H_ + h] = acc * inv;                 // context
    xbuf[(size_t)b * (2 * H_) + h] = embedded[(size_t)(b * T_ + t) * H_ + h]; // emb
  }
}

// GRU gate fusion: consumes gi (incl. b_ih) and gh (incl. b_hh)
__global__ void gru_gate(const float* __restrict__ gi, const float* __restrict__ gh,
                         float* __restrict__ h_cur, float* __restrict__ hall, int t) {
  int idx = blockIdx.x * blockDim.x + threadIdx.x;   // B*H = 32768
  if (idx >= B_ * H_) return;
  int b = idx >> 9, hh = idx & (H_ - 1);
  size_t base = (size_t)b * (3 * H_);
  float i_r = gi[base + hh], i_z = gi[base + H_ + hh], i_n = gi[base + 2 * H_ + hh];
  float h_r = gh[base + hh], h_z = gh[base + H_ + hh], h_n = gh[base + 2 * H_ + hh];
  float r = 1.0f / (1.0f + __expf(-(i_r + h_r)));
  float z = 1.0f / (1.0f + __expf(-(i_z + h_z)));
  float n = tanhf(i_n + r * h_n);
  float hn = (1.0f - z) * n + z * h_cur[idx];
  h_cur[idx] = hn;
  hall[(size_t)t * (B_ * H_) + idx] = hn;            // row = t*B + b
}

// In-place log_softmax over V per (b,t) row
__global__ void logsoftmax_kernel(float* __restrict__ lp) {
  int row = blockIdx.x;
  float* p = lp + (size_t)row * V_;
  __shared__ float sh[256];
  int tid = threadIdx.x;

  float mx = -3.4e38f;
  for (int i = tid; i < V_; i += 256) mx = fmaxf(mx, p[i]);
  sh[tid] = mx; __syncthreads();
  for (int s = 128; s; s >>= 1) { if (tid < s) sh[tid] = fmaxf(sh[tid], sh[tid + s]); __syncthreads(); }
  mx = sh[0]; __syncthreads();

  float sum = 0.0f;
  for (int i = tid; i < V_; i += 256) sum += __expf(p[i] - mx);
  sh[tid] = sum; __syncthreads();
  for (int s = 128; s; s >>= 1) { if (tid < s) sh[tid] += sh[tid + s]; __syncthreads(); }
  float lse = mx + __logf(sh[0]);
  __syncthreads();

  for (int i = tid; i < V_; i += 256) p[i] -= lse;
}

// ---------------------------------------------------------------------------
// Host-side orchestration
// ---------------------------------------------------------------------------
extern "C" void kernel_launch(void* const* d_in, const int* in_sizes, int n_in,
                              void* d_out, int out_size, void* d_ws, size_t ws_size,
                              hipStream_t stream) {
  const float* enc   = (const float*)d_in[0];   // (B,S,H)
  const float* enc_h = (const float*)d_in[1];   // (1,B,H)
  const int*   tgt   = (const int*)  d_in[2];   // (B,T)
  const float* emb   = (const float*)d_in[3];   // (V,H)
  const float* Wa    = (const float*)d_in[4];
  const float* ba    = (const float*)d_in[5];
  const float* Ua    = (const float*)d_in[6];
  const float* bu    = (const float*)d_in[7];
  const float* va    = (const float*)d_in[8];
  const float* bv    = (const float*)d_in[9];
  const float* w_ih  = (const float*)d_in[10];  // (3H,2H)
  const float* b_ih  = (const float*)d_in[11];
  const float* w_hh  = (const float*)d_in[12];  // (3H,H)
  const float* b_hh  = (const float*)d_in[13];
  const float* Wo    = (const float*)d_in[14];  // (V,H)
  const float* bo    = (const float*)d_in[15];

  float* out = (float*)d_out;
  float* out_logp = out;                                   // B*T*V
  float* out_hid  = out + (size_t)B_ * T_ * V_;            // B*H
  float* out_attn = out_hid + (size_t)B_ * H_;             // B*T*S

  char* wsb = (char*)d_ws;
  size_t off = 0;
  auto carve = [&](size_t bytes) -> void* {
    void* p = wsb + off; off += (bytes + 255) & ~(size_t)255; return p;
  };
  __bf16* Wo_bf   = (__bf16*)carve((size_t)V_ * H_ * 2);
  __bf16* Ua_bf   = (__bf16*)carve((size_t)H_ * H_ * 2);
  __bf16* Wa_bf   = (__bf16*)carve((size_t)H_ * H_ * 2);
  __bf16* wih_bf  = (__bf16*)carve((size_t)3 * H_ * 2 * H_ * 2);
  __bf16* whh_bf  = (__bf16*)carve((size_t)3 * H_ * H_ * 2);
  __bf16* hall_bf = (__bf16*)carve((size_t)T_ * B_ * H_ * 2);
  float*  ua_keys = (float*)carve((size_t)B_ * S_ * H_ * 4);
  float*  qbuf    = (float*)carve((size_t)B_ * H_ * 4);
  float*  xbuf    = (float*)carve((size_t)B_ * 2 * H_ * 4);
  float*  gi      = (float*)carve((size_t)B_ * 3 * H_ * 4);
  float*  gh      = (float*)carve((size_t)B_ * 3 * H_ * 4);
  float*  h_cur   = (float*)carve((size_t)B_ * H_ * 4);
  float*  hall    = (float*)carve((size_t)T_ * B_ * H_ * 4);
  float*  embd    = (float*)carve((size_t)B_ * T_ * H_ * 4);

  // ---- prep: bf16 weight conversion, embedding gather, h0 ----
  f32_to_bf16<<<(V_ * H_ + 255) / 256, 256, 0, stream>>>(Wo, Wo_bf, V_ * H_);
  f32_to_bf16<<<(H_ * H_ + 255) / 256, 256, 0, stream>>>(Ua, Ua_bf, H_ * H_);
  f32_to_bf16<<<(H_ * H_ + 255) / 256, 256, 0, stream>>>(Wa, Wa_bf, H_ * H_);
  f32_to_bf16<<<(3 * H_ * 2 * H_ + 255) / 256, 256, 0, stream>>>(w_ih, wih_bf, 3 * H_ * 2 * H_);
  f32_to_bf16<<<(3 * H_ * H_ + 255) / 256, 256, 0, stream>>>(w_hh, whh_bf, 3 * H_ * H_);
  gather_emb<<<B_ * T_, 256, 0, stream>>>(tgt, emb, embd);
  copy_f32<<<(B_ * H_ + 255) / 256, 256, 0, stream>>>(enc_h, h_cur, B_ * H_);

  // ---- Ua_keys = enc @ Ua^T + bu : 4096x512x512 ----
  {
    int tiles = (B_ * S_ / 16) * (H_ / 16);   // 8192
    gemm_bf16w<<<(tiles + 7) / 8, 256, 0, stream>>>(enc, Ua_bf, bu, ua_keys,
                                                    B_ * S_, H_, H_);
  }

  // ---- sequential decoder steps (~13% of FLOPs) ----
  for (int t = 0; t < T_; ++t) {
    int qt = (B_ / 16) * (H_ / 16);           // 128 tiles
    gemm_bf16w<<<(qt + 7) / 8, 256, 0, stream>>>(h_cur, Wa_bf, ba, qbuf, B_, H_, H_);
    attn_kernel<<<B_, 256, 0, stream>>>(qbuf, ua_keys, va, bv, enc, embd,
                                        xbuf, out_attn, t);
    int gt = (B_ / 16) * (3 * H_ / 16);       // 384 tiles
    gemm_bf16w<<<(gt + 7) / 8, 256, 0, stream>>>(xbuf, wih_bf, b_ih, gi, B_, 3 * H_, 2 * H_);
    gemm_bf16w<<<(gt + 7) / 8, 256, 0, stream>>>(h_cur, whh_bf, b_hh, gh, B_, 3 * H_, H_);
    gru_gate<<<(B_ * H_ + 255) / 256, 256, 0, stream>>>(gi, gh, h_cur, hall, t);
  }

  // ---- convert Hall to bf16 for the big GEMM's A operand ----
  f32_to_bf16<<<(T_ * B_ * H_ + 255) / 256, 256, 0, stream>>>(hall, hall_bf, T_ * B_ * H_);

  // ---- hoisted logits GEMM: 2048 x 32000 x 512 (~84% of FLOPs) ----
  logits_kernel<<<128 * 50, 256, 0, stream>>>(hall_bf, Wo_bf, bo, out_logp);

  // ---- log_softmax over V, in place ----
  logsoftmax_kernel<<<B_ * T_, 256, 0, stream>>>(out_logp);

  // ---- decoder_hidden = h_final ----
  copy_f32<<<(B_ * H_ + 255) / 256, 256, 0, stream>>>(h_cur, out_hid, B_ * H_);
}
